// TransformerBlock_19439021981717
// MI455X (gfx1250) — compile-verified, matrix operand order checked by reading, not run on hardware
//
#include <hip/hip_runtime.h>
#include <hip/hip_bf16.h>
#include <math.h>

typedef __attribute__((ext_vector_type(16))) _Float16 v16h;
typedef __attribute__((ext_vector_type(8)))  float    v8f;

static constexpr int E_  = 1024;
static constexpr int T_  = 1024;
static constexpr int B_  = 4;
static constexpr int H_  = 16;
static constexpr int D_  = 64;
static constexpr int F_  = 4096;

enum { EPI_F32 = 0, EPI_F16 = 1, EPI_F16_GELU = 2, EPI_F32_RES = 3 };

// ---- CDNA5 primitives -------------------------------------------------------
__device__ __forceinline__ v8f wmma_f16(v16h a, v16h b, v8f c) {
  return __builtin_amdgcn_wmma_f32_16x16x32_f16(false, a, false, b, (short)0, c,
                                                false, false);
}

// Async copy 16B/lane global -> LDS (ASYNCcnt-tracked, bypasses VGPRs).
// Per ISA 10.2, low 32 bits of a generic LDS pointer are the LDS byte address.
__device__ __forceinline__ void async_b128(const void* lds, const void* g) {
  asm volatile("global_load_async_to_lds_b128 %0, %1, off"
               :: "v"((uint32_t)(uintptr_t)lds),
                  "v"((uint64_t)(uintptr_t)g)
               : "memory");
}
__device__ __forceinline__ void wait_async0() {
  asm volatile("s_wait_asynccnt 0" ::: "memory");
}

// 16-bit A/B fragment, row-major source: element [row][k] = base[row*stride+k]
// ISA 7.12.2: lanes 0-15/16-31 hold rows 0..15, VGPR v holds
// K = (v/4)*16 + half*8 + (v%4)*2 (+k0).
__device__ __forceinline__ v16h frag_rm(const _Float16* base, int stride,
                                        int lane, int k0) {
  int m = lane & 15, half = lane >> 4;
  const _Float16* p = base + (size_t)m * stride + k0 + (half << 3);
  v16h f;
#pragma unroll
  for (int v = 0; v < 8; ++v) {
    int k = ((v >> 2) << 4) + ((v & 3) << 1);
    f[2 * v]     = p[k];
    f[2 * v + 1] = p[k + 1];
  }
  return f;
}

// B fragment from K-major source: element B[k][n] = base[k*stride + n]
__device__ __forceinline__ v16h frag_cm(const _Float16* base, int stride,
                                        int lane) {
  int n = lane & 15, half = lane >> 4;
  const _Float16* p = base + (size_t)(half << 3) * stride + n;
  v16h f;
#pragma unroll
  for (int v = 0; v < 8; ++v) {
    int k = ((v >> 2) << 4) + ((v & 3) << 1);
    f[2 * v]     = p[(size_t)k * stride];
    f[2 * v + 1] = p[(size_t)(k + 1) * stride];
  }
  return f;
}

// ---- f32 -> f16 weight conversion (one-shot, bandwidth-trivial) -------------
__global__ __launch_bounds__(256) void cvt_f16_kernel(
    const float* __restrict__ in, _Float16* __restrict__ out) {
  int i = (blockIdx.x * 256 + threadIdx.x) * 4;
  float4 v = *(const float4*)(in + i);
  out[i + 0] = (_Float16)v.x;
  out[i + 1] = (_Float16)v.y;
  out[i + 2] = (_Float16)v.z;
  out[i + 3] = (_Float16)v.w;
}

// ---- LayerNorm (f32 in -> f16 out), one row (E=1024) per 256-thread block ---
__global__ __launch_bounds__(256) void ln_f16_kernel(
    const float* __restrict__ x, const float* __restrict__ w,
    const float* __restrict__ bb, _Float16* __restrict__ out) {
  int row = blockIdx.x;
  const float* xr = x + (size_t)row * E_;
  float vals[4];
  float s = 0.f, s2 = 0.f;
#pragma unroll
  for (int i = 0; i < 4; ++i) {
    float v = xr[threadIdx.x + i * 256];
    vals[i] = v;
    s += v;
    s2 += v * v;
  }
#pragma unroll
  for (int msk = 1; msk < 32; msk <<= 1) {
    s  += __shfl_xor(s, msk, 32);
    s2 += __shfl_xor(s2, msk, 32);
  }
  __shared__ float ps[8], ps2[8];
  int lane = threadIdx.x & 31, wid = threadIdx.x >> 5;
  if (lane == 0) { ps[wid] = s; ps2[wid] = s2; }
  __syncthreads();
  s = 0.f; s2 = 0.f;
#pragma unroll
  for (int i = 0; i < 8; ++i) { s += ps[i]; s2 += ps2[i]; }
  float mean = s * (1.f / E_);
  float var  = s2 * (1.f / E_) - mean * mean;
  float rstd = rsqrtf(var + 1e-5f);
#pragma unroll
  for (int i = 0; i < 4; ++i) {
    int c = threadIdx.x + i * 256;
    out[(size_t)row * E_ + c] = (_Float16)((vals[i] - mean) * rstd * w[c] + bb[c]);
  }
}

// ---- Tiled WMMA GEMM:  out[M,N] = A[M,K] * W[N,K]^T + bias (+epilogue) ------
// A,W f16. 128x128 tile/block, 8 waves (4xM, 2xN), 32x64 per wave.
// Double-buffered LDS, tiles filled via async-to-LDS (ASYNCcnt).
template <int EPI>
__global__ __launch_bounds__(256) void gemm_wmma_kernel(
    const _Float16* __restrict__ A, const _Float16* __restrict__ Wt,
    const float* __restrict__ bias, const float* __restrict__ resid,
    void* __restrict__ out, int M, int N, int K) {
  constexpr int TK = 32;
  __shared__ __align__(16) _Float16 As[2][128 * TK];
  __shared__ __align__(16) _Float16 Bs[2][128 * TK];
  int tid = threadIdx.x, lane = tid & 31, wid = tid >> 5;
  int wm = wid & 3, wn = wid >> 2;
  int bm = blockIdx.y * 128, bn = blockIdx.x * 128;
  int ldrow = tid >> 1, ldcol = (tid & 1) << 4;  // 128 rows x 32 k, 16 h/thread
  const _Float16* ag = A  + (size_t)(bm + ldrow) * K + ldcol;
  const _Float16* wg = Wt + (size_t)(bn + ldrow) * K + ldcol;
  const int iters = K / TK;

  // prefetch tile 0 into buffer 0
  {
    _Float16* la = &As[0][ldrow * TK + ldcol];
    _Float16* lb = &Bs[0][ldrow * TK + ldcol];
    async_b128(la, ag);     async_b128(la + 8, ag + 8);
    async_b128(lb, wg);     async_b128(lb + 8, wg + 8);
  }

  v8f acc[2][4] = {};
  for (int it = 0; it < iters; ++it) {
    int cur = it & 1;
    wait_async0();        // my in-flight ops are exactly buffer `cur`'s
    __syncthreads();      // all waves done writing `cur` & done reading `cur^1`
    if (it + 1 < iters) { // prefetch next tile into the other buffer
      const _Float16* ga = ag + (size_t)(it + 1) * TK;
      const _Float16* gb = wg + (size_t)(it + 1) * TK;
      _Float16* la = &As[cur ^ 1][ldrow * TK + ldcol];
      _Float16* lb = &Bs[cur ^ 1][ldrow * TK + ldcol];
      async_b128(la, ga);  async_b128(la + 8, ga + 8);
      async_b128(lb, gb);  async_b128(lb + 8, gb + 8);
    }
    v16h af0 = frag_rm(&As[cur][(wm * 32) * TK], TK, lane, 0);
    v16h af1 = frag_rm(&As[cur][(wm * 32 + 16) * TK], TK, lane, 0);
#pragma unroll
    for (int j = 0; j < 4; ++j) {
      v16h bf = frag_rm(&Bs[cur][(wn * 64 + j * 16) * TK], TK, lane, 0);
      acc[0][j] = wmma_f16(af0, bf, acc[0][j]);
      acc[1][j] = wmma_f16(af1, bf, acc[1][j]);
    }
    __syncthreads();      // all waves done reading `cur` before it is refilled
  }

  int n16 = lane & 15, half = lane >> 4;
#pragma unroll
  for (int i = 0; i < 2; ++i)
#pragma unroll
    for (int j = 0; j < 4; ++j) {
      int col = bn + wn * 64 + j * 16 + n16;
      float bv = bias[col];
#pragma unroll
      for (int r = 0; r < 8; ++r) {
        int row = bm + wm * 32 + i * 16 + half * 8 + r;
        size_t idx = (size_t)row * N + col;
        float v = acc[i][j][r] + bv;
        if constexpr (EPI == EPI_F32) {
          ((float*)out)[idx] = v;
        } else if constexpr (EPI == EPI_F16) {
          ((_Float16*)out)[idx] = (_Float16)v;
        } else if constexpr (EPI == EPI_F16_GELU) {
          ((_Float16*)out)[idx] =
              (_Float16)(0.5f * v * (1.f + erff(v * 0.70710678118f)));
        } else {
          ((float*)out)[idx] = v + resid[idx];
        }
      }
    }
}

// ---- Flash-style causal attention -------------------------------------------
// grid: (T/128 row blocks, B*H). 8 waves/block, each wave owns 16 query rows.
__global__ __launch_bounds__(256) void attn_kernel(
    const _Float16* __restrict__ qkv, _Float16* __restrict__ o16) {
  __shared__ __align__(16) _Float16 Pls[8 * 16 * 32];  // per-wave P staging
  int tid = threadIdx.x, lane = tid & 31, wid = tid >> 5;
  int b = blockIdx.y >> 4, h = blockIdx.y & 15;
  int q0 = blockIdx.x * 128 + wid * 16;
  const int RS = 3 * E_;
  const _Float16* qbase = qkv + (size_t)b * T_ * RS + h * D_;
  const _Float16* kbase = qbase + E_;
  const _Float16* vbase = qbase + 2 * E_;
  int n16 = lane & 15, half = lane >> 4;

  v16h qf0 = frag_rm(qbase + (size_t)q0 * RS, RS, lane, 0);
  v16h qf1 = frag_rm(qbase + (size_t)q0 * RS, RS, lane, 32);
  v8f oacc[4] = {};
  float mrow[8], lrow[8];
#pragma unroll
  for (int r = 0; r < 8; ++r) { mrow[r] = -INFINITY; lrow[r] = 0.f; }
  _Float16* Pw = &Pls[wid * 16 * 32];
  const float SC = 0.125f;  // 1/sqrt(64)

  for (int j0 = 0; j0 <= q0 + 15; j0 += 32) {   // wave-uniform causal bound
    v8f s0 = {}, s1 = {};
    s0 = wmma_f16(qf0, frag_rm(kbase + (size_t)j0 * RS, RS, lane, 0), s0);
    s0 = wmma_f16(qf1, frag_rm(kbase + (size_t)j0 * RS, RS, lane, 32), s0);
    s1 = wmma_f16(qf0, frag_rm(kbase + (size_t)(j0 + 16) * RS, RS, lane, 0), s1);
    s1 = wmma_f16(qf1, frag_rm(kbase + (size_t)(j0 + 16) * RS, RS, lane, 32), s1);
#pragma unroll
    for (int r = 0; r < 8; ++r) {
      int qrow = q0 + half * 8 + r;
      float a = s0[r] * SC; if (j0 + n16      > qrow) a = -INFINITY;
      float c = s1[r] * SC; if (j0 + 16 + n16 > qrow) c = -INFINITY;
      float t = fmaxf(a, c);
#pragma unroll
      for (int msk = 1; msk < 16; msk <<= 1)
        t = fmaxf(t, __shfl_xor(t, msk, 32));  // row max over 16-lane half
      float mn    = fmaxf(mrow[r], t);
      float alpha = __expf(mrow[r] - mn);
      float p0    = __expf(a - mn);
      float p1    = __expf(c - mn);
      float rs    = p0 + p1;
#pragma unroll
      for (int msk = 1; msk < 16; msk <<= 1) rs += __shfl_xor(rs, msk, 32);
      lrow[r] = lrow[r] * alpha + rs;
      mrow[r] = mn;
#pragma unroll
      for (int d = 0; d < 4; ++d) oacc[d][r] *= alpha;
      Pw[(half * 8 + r) * 32 + n16]      = (_Float16)p0;
      Pw[(half * 8 + r) * 32 + 16 + n16] = (_Float16)p1;
    }
    v16h pf = frag_rm(Pw, 32, lane, 0);
#pragma unroll
    for (int d = 0; d < 4; ++d)
      oacc[d] = wmma_f16(pf, frag_cm(vbase + (size_t)j0 * RS + d * 16, RS, lane),
                         oacc[d]);
  }
  _Float16* obase = o16 + (size_t)b * T_ * E_ + h * D_;
#pragma unroll
  for (int r = 0; r < 8; ++r) {
    float inv = 1.0f / lrow[r];
    int trow = q0 + half * 8 + r;
#pragma unroll
    for (int d = 0; d < 4; ++d)
      obase[(size_t)trow * E_ + d * 16 + n16] = (_Float16)(oacc[d][r] * inv);
  }
}

// ---- Launch -----------------------------------------------------------------
extern "C" void kernel_launch(void* const* d_in, const int* in_sizes, int n_in,
                              void* d_out, int out_size, void* d_ws,
                              size_t ws_size, hipStream_t stream) {
  (void)in_sizes; (void)n_in; (void)out_size; (void)ws_size;
  const float* x    = (const float*)d_in[0];
  const float* ln1w = (const float*)d_in[1];
  const float* ln1b = (const float*)d_in[2];
  const float* ln2w = (const float*)d_in[3];
  const float* ln2b = (const float*)d_in[4];
  const float* qkvw = (const float*)d_in[5];
  const float* qkvb = (const float*)d_in[6];
  const float* outw = (const float*)d_in[7];
  const float* outb = (const float*)d_in[8];
  const float* fc1w = (const float*)d_in[9];
  const float* fc1b = (const float*)d_in[10];
  const float* fc2w = (const float*)d_in[11];
  const float* fc2b = (const float*)d_in[12];

  const int M = B_ * T_;  // 4096
  char* ws = (char*)d_ws;
  // [0,24MB): f16 weights; buf0 8MB; buf1 32MB; x1 16MB  (total 80MB)
  _Float16* qkvw16 = (_Float16*)(ws);
  _Float16* outw16 = (_Float16*)(ws + (6ull  << 20));
  _Float16* fc1w16 = (_Float16*)(ws + (8ull  << 20));
  _Float16* fc2w16 = (_Float16*)(ws + (16ull << 20));
  _Float16* buf0   = (_Float16*)(ws + (24ull << 20));
  _Float16* buf1   = (_Float16*)(ws + (32ull << 20));
  float*    x1     = (float*)   (ws + (64ull << 20));

  dim3 blk(256);
  cvt_f16_kernel<<<dim3(3 * E_ * E_ / 1024), blk, 0, stream>>>(qkvw, qkvw16);
  cvt_f16_kernel<<<dim3(E_ * E_ / 1024),     blk, 0, stream>>>(outw, outw16);
  cvt_f16_kernel<<<dim3(F_ * E_ / 1024),     blk, 0, stream>>>(fc1w, fc1w16);
  cvt_f16_kernel<<<dim3(E_ * F_ / 1024),     blk, 0, stream>>>(fc2w, fc2w16);

  ln_f16_kernel<<<dim3(M), blk, 0, stream>>>(x, ln1w, ln1b, buf0);
  gemm_wmma_kernel<EPI_F16><<<dim3(3 * E_ / 128, M / 128), blk, 0, stream>>>(
      buf0, qkvw16, qkvb, nullptr, buf1, M, 3 * E_, E_);
  attn_kernel<<<dim3(T_ / 128, B_ * H_), blk, 0, stream>>>(buf1, buf0);
  gemm_wmma_kernel<EPI_F32_RES><<<dim3(E_ / 128, M / 128), blk, 0, stream>>>(
      buf0, outw16, outb, x, x1, M, E_, E_);
  ln_f16_kernel<<<dim3(M), blk, 0, stream>>>(x1, ln2w, ln2b, buf0);
  gemm_wmma_kernel<EPI_F16_GELU><<<dim3(F_ / 128, M / 128), blk, 0, stream>>>(
      buf0, fc1w16, fc1b, nullptr, buf1, M, F_, E_);
  gemm_wmma_kernel<EPI_F32_RES><<<dim3(E_ / 128, M / 128), blk, 0, stream>>>(
      buf1, fc2w16, fc2b, x1, d_out, M, E_, F_);
}
